// MultiHeadAttention_64965675320045
// MI455X (gfx1250) — matrix operand order repaired and checked by reading, not compiled, not run on hardware
//
#include <hip/hip_runtime.h>
#include <hip/hip_bf16.h>

typedef __attribute__((ext_vector_type(16))) __bf16 v16bf;
typedef __attribute__((ext_vector_type(8)))  __bf16 v8bf;
typedef __attribute__((ext_vector_type(8)))  float  v8f;

#define BB 2
#define SS 4096
#define DD 512
#define HH 8
#define HDIM 64

#define WMMA_BF16(a,b,c) __builtin_amdgcn_wmma_f32_16x16x32_bf16(false,(a),false,(b),(short)0,(c),false,false)

// ---- CDNA5 async global->LDS (ASYNCcnt) -----------------------------------
// LDS byte offset = low 32 bits of the flat address (addrspacecast local->flat
// is {aperture_hi, offset32}; aperture map truncates to addr[31:0]).
static __device__ __forceinline__ unsigned lds_addr32(const void* p) {
  return (unsigned)(unsigned long long)p;
}
static __device__ __forceinline__ void async_load_lds_b128(unsigned lds, const void* g) {
  asm volatile("global_load_async_to_lds_b128 %0, %1, off"
               :: "v"(lds), "v"((unsigned long long)g) : "memory");
}
static __device__ __forceinline__ void wait_async0() {
  asm volatile("s_wait_asynccnt 0x0" ::: "memory");
}

// A-fragment (16x32 bf16, MxK) from LDS row-major [m][k], row stride ld elems.
// Lane l: m=l%16, h=l/16. Elems 0..7 -> k=h*8+e ; 8..15 -> k=16+h*8+(e-8).
static __device__ __forceinline__ v16bf lds_load_a(const __bf16* base, int ld) {
  int lane = threadIdx.x & 31;
  int m = lane & 15, h = lane >> 4;
  const __bf16* p0 = base + m * ld + h * 8;
  v8bf lo = *(const v8bf*)p0;
  v8bf hi = *(const v8bf*)(p0 + 16);
  v16bf r;
#pragma unroll
  for (int i = 0; i < 8; ++i) { r[i] = lo[i]; r[8 + i] = hi[i]; }
  return r;
}

// B-fragment (32x16 bf16, KxN) from LDS stored [n][k], row stride ld elems.
// Lane l: n=l%16, h=l/16. Elems 0..7 -> k=h*8+e ; 8..15 -> k=16+h*8+(e-8).
static __device__ __forceinline__ v16bf lds_load_b(const __bf16* base, int ld) {
  int lane = threadIdx.x & 31;
  int n = lane & 15, h = lane >> 4;
  const __bf16* p = base + n * ld + h * 8;
  v8bf lo = *(const v8bf*)p;
  v8bf hi = *(const v8bf*)(p + 16);
  v16bf r;
#pragma unroll
  for (int i = 0; i < 8; ++i) { r[i] = lo[i]; r[8 + i] = hi[i]; }
  return r;
}

// ---------------------------------------------------------------------------
// GEMM: out[M,N] = A[M,K](f32) * W[K,N](f32) + bias[N].
// 128x64 tile per block, 256 threads = 8 waves; wave -> 32x32 (4 WMMA/step).
// ---------------------------------------------------------------------------
template<bool STORE_BF16>
__global__ __launch_bounds__(256) void gemm_bias_kernel(
    const float* __restrict__ A, const float* __restrict__ W,
    const float* __restrict__ bias, void* __restrict__ outp,
    int M, int N, int K) {
  __shared__ __bf16 Asl[128][40];   // [m][k]
  __shared__ __bf16 Bsl[64][40];    // [n][k] (transposed W tile)

  const int t = threadIdx.x, lane = t & 31, w = t >> 5;
  const int m0 = blockIdx.y * 128, n0 = blockIdx.x * 64;
  const int m_off = (w >> 1) * 32, n_off = (w & 1) * 32;

  v8f acc00 = {}, acc01 = {}, acc10 = {}, acc11 = {};

  for (int ko = 0; ko < K; ko += 32) {
    __syncthreads();
    { // A tile 128x32: thread -> row t/2, 16 cols at (t%2)*16
      int r = t >> 1, c = (t & 1) * 16;
      const float4* src = (const float4*)(A + (size_t)(m0 + r) * K + ko + c);
      float4 f0 = src[0], f1 = src[1], f2 = src[2], f3 = src[3];
      __bf16* dst = &Asl[r][c];
      dst[0]  = (__bf16)f0.x; dst[1]  = (__bf16)f0.y; dst[2]  = (__bf16)f0.z; dst[3]  = (__bf16)f0.w;
      dst[4]  = (__bf16)f1.x; dst[5]  = (__bf16)f1.y; dst[6]  = (__bf16)f1.z; dst[7]  = (__bf16)f1.w;
      dst[8]  = (__bf16)f2.x; dst[9]  = (__bf16)f2.y; dst[10] = (__bf16)f2.z; dst[11] = (__bf16)f2.w;
      dst[12] = (__bf16)f3.x; dst[13] = (__bf16)f3.y; dst[14] = (__bf16)f3.z; dst[15] = (__bf16)f3.w;
    }
    { // W tile 32x64 -> transposed into Bsl[n][k]
      int kr = t >> 3, c = (t & 7) * 8;
      const float4* src = (const float4*)(W + (size_t)(ko + kr) * N + n0 + c);
      float4 f0 = src[0], f1 = src[1];
      Bsl[c + 0][kr] = (__bf16)f0.x; Bsl[c + 1][kr] = (__bf16)f0.y;
      Bsl[c + 2][kr] = (__bf16)f0.z; Bsl[c + 3][kr] = (__bf16)f0.w;
      Bsl[c + 4][kr] = (__bf16)f1.x; Bsl[c + 5][kr] = (__bf16)f1.y;
      Bsl[c + 6][kr] = (__bf16)f1.z; Bsl[c + 7][kr] = (__bf16)f1.w;
    }
    __syncthreads();

    v16bf a0 = lds_load_a(&Asl[m_off][0], 40);
    v16bf a1 = lds_load_a(&Asl[m_off + 16][0], 40);
    v16bf b0 = lds_load_b(&Bsl[n_off][0], 40);
    v16bf b1 = lds_load_b(&Bsl[n_off + 16][0], 40);
    acc00 = WMMA_BF16(a0, b0, acc00);
    acc01 = WMMA_BF16(a0, b1, acc01);
    acc10 = WMMA_BF16(a1, b0, acc10);
    acc11 = WMMA_BF16(a1, b1, acc11);
  }

  { // Epilogue: VGPR r -> rows r / r+8 (lane>=16), col = lane%16.
    int n = lane & 15, hh = lane >> 4;
    int gn = n0 + n_off + n;
    float bn0 = bias[gn], bn1 = bias[gn + 16];
#pragma unroll
    for (int r = 0; r < 8; ++r) {
      size_t row0 = (size_t)(m0 + m_off + hh * 8 + r) * N;
      size_t row1 = row0 + (size_t)16 * N;
      float v00 = acc00[r] + bn0, v01 = acc01[r] + bn1;
      float v10 = acc10[r] + bn0, v11 = acc11[r] + bn1;
      if (STORE_BF16) {
        __bf16* ob = (__bf16*)outp;
        ob[row0 + gn] = (__bf16)v00; ob[row0 + gn + 16] = (__bf16)v01;
        ob[row1 + gn] = (__bf16)v10; ob[row1 + gn + 16] = (__bf16)v11;
      } else {
        float* of = (float*)outp;
        of[row0 + gn] = v00; of[row0 + gn + 16] = v01;
        of[row1 + gn] = v10; of[row1 + gn + 16] = v11;
      }
    }
  }
}

// ---------------------------------------------------------------------------
// Flash attention: block = (b,h, 64-query tile), 256 threads = 8 waves.
// qkv: bf16 [B*S][3*D], per-token layout [H][3*HD] (q|k|v per head).
// attn_out: f32 [B*S][D] (head-concatenated).
// ---------------------------------------------------------------------------
__global__ __launch_bounds__(256) void attn_kernel(
    const __bf16* __restrict__ qkv, const float* __restrict__ mask,
    float* __restrict__ attn_out) {
  __shared__ __bf16 Qs[64][72];   // [q][hd]
  __shared__ __bf16 Ks[64][72];   // [key][hd]  == B-frag layout for Q*K^T
  __shared__ __bf16 Vt[64][72];   // [hd][key]  == B-frag layout for P*V
  __shared__ __bf16 Pb[64][72];   // [q][key]   softmax probs (bf16)
  __shared__ float  Sc[64][65];   // raw scores (f32)
  __shared__ float  rowM[64], rowL[64], rowScale[64];

  const int t = threadIdx.x, lane = t & 31, w = t >> 5;
  const int bh = blockIdx.y, b = bh / HH, h = bh % HH;
  const int q0 = blockIdx.x * 64;
  const size_t rowStride = 3 * DD;            // 1536
  const size_t baseTok = (size_t)b * SS;
  const int colQ = h * 3 * HDIM;
  const int colK = colQ + HDIM;
  const int colV = colQ + 2 * HDIM;

  const int r = t >> 2, cseg = (t & 3) * 16;  // tile-staging mapping (64x64)

  { // stage Q tile via async copy (natural layout, already bf16)
    const __bf16* src = qkv + (baseTok + q0 + r) * rowStride + colQ + cseg;
    unsigned dst = lds_addr32(&Qs[r][cseg]);
    async_load_lds_b128(dst, src);
    async_load_lds_b128(dst + 16, src + 8);
  }
  if (t < 64) { rowM[t] = -1e30f; rowL[t] = 0.f; rowScale[t] = 1.f; }
  wait_async0();
  __syncthreads();

  const int m_off = (w >> 1) * 16, n_off = (w & 1) * 32;
  const v16bf aq0 = lds_load_a(&Qs[m_off][0], 72);    // hd 0..31
  const v16bf aq1 = lds_load_a(&Qs[m_off][32], 72);   // hd 32..63
  v8f o0 = {}, o1 = {};
  const float scale = 0.125f;  // 1/sqrt(64)

  for (int kt = 0; kt < SS / 64; ++kt) {
    const int kbase = kt * 64;
    __syncthreads();
    { // K tile: async straight to LDS (natural layout)
      const __bf16* srcK = qkv + (baseTok + kbase + r) * rowStride + colK + cseg;
      unsigned dstK = lds_addr32(&Ks[r][cseg]);
      async_load_lds_b128(dstK, srcK);
      async_load_lds_b128(dstK + 16, srcK + 8);
    }
    { // V tile: needs transpose -> stage through VGPRs
      const __bf16* srcV = qkv + (baseTok + kbase + r) * rowStride + colV + cseg;
      v8bf v0v = *(const v8bf*)srcV;
      v8bf v1v = *(const v8bf*)(srcV + 8);
#pragma unroll
      for (int i = 0; i < 8; ++i) { Vt[cseg + i][r] = v0v[i]; Vt[cseg + 8 + i][r] = v1v[i]; }
    }
    wait_async0();
    __syncthreads();

    // scores = Q @ K^T  (64x64, each wave 16x32)
    v8f s0 = {}, s1 = {};
    {
      v16bf bk00 = lds_load_b(&Ks[n_off][0], 72);
      v16bf bk01 = lds_load_b(&Ks[n_off][32], 72);
      v16bf bk10 = lds_load_b(&Ks[n_off + 16][0], 72);
      v16bf bk11 = lds_load_b(&Ks[n_off + 16][32], 72);
      s0 = WMMA_BF16(aq0, bk00, s0);
      s0 = WMMA_BF16(aq1, bk01, s0);
      s1 = WMMA_BF16(aq0, bk10, s1);
      s1 = WMMA_BF16(aq1, bk11, s1);
    }
    {
      int n = lane & 15, hh = lane >> 4;
#pragma unroll
      for (int rr = 0; rr < 8; ++rr) {
        int m = m_off + hh * 8 + rr;
        Sc[m][n_off + n]      = s0[rr] * scale;
        Sc[m][n_off + 16 + n] = s1[rr] * scale;
      }
    }
    __syncthreads();

    // online softmax over this key tile: thread -> row t/4, 16 cols
    {
      const float* mrow = mask + (size_t)(q0 + r) * SS + kbase + cseg;
      float vbuf[16];
      float mx = -1e30f;
#pragma unroll
      for (int i = 0; i < 16; ++i) {
        float v = Sc[r][cseg + i] + mrow[i];
        vbuf[i] = v;
        mx = fmaxf(mx, v);
      }
      mx = fmaxf(mx, __shfl_xor(mx, 1, 32));
      mx = fmaxf(mx, __shfl_xor(mx, 2, 32));
      float newM = fmaxf(rowM[r], mx);
      float sum = 0.f;
#pragma unroll
      for (int i = 0; i < 16; ++i) {
        float p = __expf(vbuf[i] - newM);
        Pb[r][cseg + i] = (__bf16)p;
        sum += p;
      }
      sum += __shfl_xor(sum, 1, 32);
      sum += __shfl_xor(sum, 2, 32);
      if ((t & 3) == 0) {
        float sc = __expf(rowM[r] - newM);
        rowL[r] = rowL[r] * sc + sum;
        rowM[r] = newM;
        rowScale[r] = sc;
      }
    }
    __syncthreads();

    // rescale O and accumulate P @ V
    {
      int hh = lane >> 4;
#pragma unroll
      for (int rr = 0; rr < 8; ++rr) {
        float sc = rowScale[m_off + hh * 8 + rr];
        o0[rr] *= sc; o1[rr] *= sc;
      }
      v16bf ap0  = lds_load_a(&Pb[m_off][0], 72);        // keys 0..31
      v16bf ap1  = lds_load_a(&Pb[m_off][32], 72);       // keys 32..63
      v16bf bv00 = lds_load_b(&Vt[n_off][0], 72);
      v16bf bv01 = lds_load_b(&Vt[n_off][32], 72);
      v16bf bv10 = lds_load_b(&Vt[n_off + 16][0], 72);
      v16bf bv11 = lds_load_b(&Vt[n_off + 16][32], 72);
      o0 = WMMA_BF16(ap0, bv00, o0);
      o0 = WMMA_BF16(ap1, bv01, o0);
      o1 = WMMA_BF16(ap0, bv10, o1);
      o1 = WMMA_BF16(ap1, bv11, o1);
    }
    // prefetch next K tile row
    if (kt + 1 < SS / 64) {
      const __bf16* nxt = qkv + (baseTok + kbase + 64 + r) * rowStride + colK;
      __builtin_prefetch(nxt, 0, 1);
    }
  }

  __syncthreads();
  { // normalize and write: attn_out[(b*S+q)*D + h*64 + n]
    int n = lane & 15, hh = lane >> 4;
#pragma unroll
    for (int rr = 0; rr < 8; ++rr) {
      int m = m_off + hh * 8 + rr;
      float invl = 1.f / rowL[m];
      size_t row = (baseTok + q0 + m) * DD + (size_t)h * HDIM;
      attn_out[row + n_off + n]      = o0[rr] * invl;
      attn_out[row + n_off + 16 + n] = o1[rr] * invl;
    }
  }
}

extern "C" void kernel_launch(void* const* d_in, const int* in_sizes, int n_in,
                              void* d_out, int out_size, void* d_ws, size_t ws_size,
                              hipStream_t stream) {
  (void)in_sizes; (void)n_in; (void)out_size; (void)ws_size;
  const float* x     = (const float*)d_in[0];
  const float* mask  = (const float*)d_in[1];
  const float* W_qkv = (const float*)d_in[2];
  const float* b_qkv = (const float*)d_in[3];
  const float* W_out = (const float*)d_in[4];
  const float* b_out = (const float*)d_in[5];
  float* out = (float*)d_out;

  __bf16* qkv = (__bf16*)d_ws;
  size_t qkvBytes = (size_t)BB * SS * 3 * DD * sizeof(__bf16);   // ~25.2 MB
  float* attn = (float*)((char*)d_ws + ((qkvBytes + 255) & ~(size_t)255));

  dim3 blk(256);
  // 1) QKV projection: [8192,512] x [512,1536] -> bf16
  gemm_bias_kernel<true><<<dim3((3 * DD) / 64, (BB * SS) / 128), blk, 0, stream>>>(
      x, W_qkv, b_qkv, (void*)qkv, BB * SS, 3 * DD, DD);
  // 2) flash attention
  attn_kernel<<<dim3(SS / 64, BB * HH), blk, 0, stream>>>(qkv, mask, attn);
  // 3) output projection: [8192,512] x [512,512] -> f32
  gemm_bias_kernel<false><<<dim3(DD / 64, (BB * SS) / 128), blk, 0, stream>>>(
      attn, W_out, b_out, (void*)out, BB * SS, DD, DD);
}
